// HomogeneousGraphSequentialClassifierV2_38860864094250
// MI455X (gfx1250) — matrix operand order repaired and checked by reading, not compiled, not run on hardware
//
#include <hip/hip_runtime.h>
#include <math.h>

// ---------------------------------------------------------------------------
// Types for CDNA5 WMMA (wave32)
// ---------------------------------------------------------------------------
typedef __attribute__((ext_vector_type(16))) _Float16 v16h;
typedef __attribute__((ext_vector_type(8)))  float    v8f;

#define WAVE 32

__device__ __forceinline__ float wave_sum(float v) {
#pragma unroll
  for (int m = 16; m >= 1; m >>= 1) v += __shfl_xor(v, m, WAVE);
  return v;
}

__device__ __forceinline__ float gelu_exact(float y) {
  return 0.5f * y * (1.0f + erff(y * 0.70710678118654752f));
}

// CDNA5 packed f16 atomic add (GLOBAL_ATOMIC_PK_ADD_F16, no return -> STOREcnt)
__device__ __forceinline__ void atomic_pk_add_f16(_Float16* addr, unsigned packed2) {
  asm volatile("global_atomic_pk_add_f16 %0, %1, off"
               :: "v"(addr), "v"(packed2)
               : "memory");
}

// ---------------------------------------------------------------------------
// WMMA fragment handling (wave32; A/B use contiguous k = 16g + s in both; C/D per CDNA5 ISA 7.12.2)
// A 16x32 f16: lane l -> row m=l&15, group g=l>>4.
//   VGPR j: K = 16g + 2j + {0,1}
// B 32x16 f16: lane l -> col n=l&15, group g=l>>4.  VGPR j: K = 16g + 2j + {0,1}
// C/D 16x16 f32: lane l -> col n=l&15; VGPR v -> row m = v + 8*(l>>4)
// ---------------------------------------------------------------------------
__device__ __forceinline__ v16h load_a_frag_lds(const _Float16* sA, int lda,
                                                int kbase, int lane) {
  const int m = lane & 15;
  const int g = lane >> 4;
  union { v16h v; unsigned u[8]; } r;
#pragma unroll
  for (int j = 0; j < 8; ++j) {
    const int k = kbase + 2 * j + 16 * g;
    r.u[j] = *(const unsigned*)(sA + m * lda + k);   // two contiguous halves
  }
  return r.v;
}

// B weights pre-swizzled to fragment-major order: lane's 16 halves contiguous.
// offset = (((kt*NT + nt)*32 + lane) << 4) halves -> 32B-aligned v16h load.
__device__ __forceinline__ v16h load_b_frag_sw(const _Float16* Wsw, int kt,
                                               int NT, int nt, int lane) {
  return *(const v16h*)(Wsw + ((((size_t)kt * NT + nt) * 32 + lane) << 4));
}

// ---------------------------------------------------------------------------
// Utility kernels
// ---------------------------------------------------------------------------
__global__ void fill_kernel(float* p, float v, int n) {
  int i = blockIdx.x * blockDim.x + threadIdx.x;
  if (i < n) p[i] = v;
}

// Swizzle f32 row-major W[K][Nw] -> f16 fragment-major (see load_b_frag_sw).
__global__ void swizzle_b_kernel(const float* __restrict__ W,
                                 _Float16* __restrict__ out, int K, int Nw) {
  int i = blockIdx.x * blockDim.x + threadIdx.x;
  if (i >= K * Nw) return;
  int h    = i & 15;
  int lane = (i >> 4) & 31;
  int tile = i >> 9;                 // kt*NT + nt
  int NT = Nw >> 4;
  int kt = tile / NT, nt = tile - kt * NT;
  int g = lane >> 4;
  int n = nt * 16 + (lane & 15);
  int k = kt * 32 + 16 * g + h;
  out[i] = (_Float16)W[k * Nw + n];
}

// ---------------------------------------------------------------------------
// 1) LayerNorm over x rows (128) -> h0 (f16).  One wave per row.
// ---------------------------------------------------------------------------
__global__ void ln_x_kernel(const float* __restrict__ x,
                            const float* __restrict__ w,
                            const float* __restrict__ b,
                            _Float16* __restrict__ h0h, int N) {
  int row = (blockIdx.x * blockDim.x + threadIdx.x) >> 5;
  int lane = threadIdx.x & 31;
  if (row >= N) return;
  const float* xr = x + (size_t)row * 128;
  float v[4], s = 0.f;
#pragma unroll
  for (int t = 0; t < 4; ++t) { v[t] = xr[lane + 32 * t]; s += v[t]; }
  s = wave_sum(s);
  float mu = s * (1.f / 128.f);
  float vs = 0.f;
#pragma unroll
  for (int t = 0; t < 4; ++t) { float d = v[t] - mu; vs += d * d; }
  vs = wave_sum(vs);
  float rstd = rsqrtf(vs * (1.f / 128.f) + 1e-5f);
  _Float16* hr = h0h + (size_t)row * 128;
#pragma unroll
  for (int t = 0; t < 4; ++t) {
    int c = lane + 32 * t;
    hr[c] = (_Float16)((v[t] - mu) * rstd * w[c] + b[c]);
  }
}

// ---------------------------------------------------------------------------
// 2) Edge scatter (f16): agg[dst] += h0[src]; deg[dst] += 1.  One wave/edge.
//    Packed-f16 atomics halve the dominant HBM/L2 traffic of the pipeline.
// ---------------------------------------------------------------------------
__global__ void scatter_kernel(const _Float16* __restrict__ h0h,
                               const int* __restrict__ src,
                               const int* __restrict__ dst,
                               _Float16* __restrict__ agg,
                               float* __restrict__ deg, int E) {
  int e = (blockIdx.x * blockDim.x + threadIdx.x) >> 5;
  int lane = threadIdx.x & 31;
  if (e >= E) return;
  int s = src[e], d = dst[e];
  uint2 v = ((const uint2*)(h0h + (size_t)s * 128))[lane];   // 4 halves
  _Float16* ap = agg + (size_t)d * 128 + lane * 4;
  atomic_pk_add_f16(ap,     v.x);
  atomic_pk_add_f16(ap + 2, v.y);
  if (lane == 0) atomicAdd(deg + d, 1.0f);
}

// ---------------------------------------------------------------------------
// 3) SAGE GEMM (fused): h1 = GELU(LN( (agg/deg)@W_l + b_l + h0@W_r ))
//    Block = 8 waves, 16 rows x 128 cols; 8x v_wmma_f32_16x16x32_f16 / wave.
// ---------------------------------------------------------------------------
#define LDA_SAGE 136   // halves; 68-DWORD stride -> conflict-free frag reads
#define LDC_SAGE 130   // floats; kills 2-way C-store conflicts
__global__ void sage_gemm_kernel(const _Float16* __restrict__ h0h,
                                 const _Float16* __restrict__ aggh,
                                 const float* __restrict__ deg,
                                 const _Float16* __restrict__ Wl_sw,
                                 const _Float16* __restrict__ Wr_sw,
                                 const float* __restrict__ b_l,
                                 const float* __restrict__ lnw,
                                 const float* __restrict__ lnb,
                                 float* __restrict__ h1, int N) {
  __shared__ _Float16 sA1[16 * LDA_SAGE];
  __shared__ _Float16 sA2[16 * LDA_SAGE];
  __shared__ float    sC[16 * LDC_SAGE];

  const int row0 = blockIdx.x * 16;
  const int tid  = threadIdx.x;

  // cooperative stage: one uint4 (8 halves) per thread per matrix
  {
    const int r  = tid >> 4;          // 0..15
    const int ch = tid & 15;          // 16B chunk within the 128-half row
    const int gr = row0 + r;
    uint4 hv = {0u, 0u, 0u, 0u};
    uint4 av = {0u, 0u, 0u, 0u};
    float inv = 1.f;
    if (gr < N) {
      hv = ((const uint4*)(h0h  + (size_t)gr * 128))[ch];
      av = ((const uint4*)(aggh + (size_t)gr * 128))[ch];
      float d = deg[gr];
      inv = 1.f / (d > 1.f ? d : 1.f);
    }
    union { uint4 u; _Float16 h[8]; } a, o;
    a.u = av;
#pragma unroll
    for (int t = 0; t < 8; ++t) o.h[t] = (_Float16)((float)a.h[t] * inv);
    *(uint4*)(sA1 + r * LDA_SAGE + ch * 8) = o.u;   // mean-neighbor tile
    *(uint4*)(sA2 + r * LDA_SAGE + ch * 8) = hv;    // self tile
  }
  __syncthreads();

  const int wave = tid >> 5, lane = tid & 31;
  v8f acc = {};
#pragma unroll
  for (int kt = 0; kt < 4; ++kt) {
    v16h a1 = load_a_frag_lds(sA1, LDA_SAGE, kt * 32, lane);
    v16h b1 = load_b_frag_sw(Wl_sw, kt, 8, wave, lane);
    acc = __builtin_amdgcn_wmma_f32_16x16x32_f16(false, a1, false, b1,
                                                 (short)0, acc, false, false);
    v16h a2 = load_a_frag_lds(sA2, LDA_SAGE, kt * 32, lane);
    v16h b2 = load_b_frag_sw(Wr_sw, kt, 8, wave, lane);
    acc = __builtin_amdgcn_wmma_f32_16x16x32_f16(false, a2, false, b2,
                                                 (short)0, acc, false, false);
  }

  // + bias, stage C into LDS for the row-wise LN epilogue
  const int col = wave * 16 + (lane & 15);
  const int grp = lane >> 4;
  const float bias = b_l[col];
#pragma unroll
  for (int v = 0; v < 8; ++v) {
    int r = v + 8 * grp;
    sC[r * LDC_SAGE + col] = acc[v] + bias;
  }
  __syncthreads();

  // LN + exact GELU: wave handles 2 rows
#pragma unroll
  for (int rr = 0; rr < 2; ++rr) {
    int r = wave * 2 + rr;
    int gr = row0 + r;
    float v[4], s = 0.f;
#pragma unroll
    for (int t = 0; t < 4; ++t) { v[t] = sC[r * LDC_SAGE + lane + 32 * t]; s += v[t]; }
    s = wave_sum(s);
    float mu = s * (1.f / 128.f);
    float vs = 0.f;
#pragma unroll
    for (int t = 0; t < 4; ++t) { float d = v[t] - mu; vs += d * d; }
    vs = wave_sum(vs);
    float rstd = rsqrtf(vs * (1.f / 128.f) + 1e-5f);
    if (gr < N) {
      float* hr = h1 + (size_t)gr * 128;
#pragma unroll
      for (int t = 0; t < 4; ++t) {
        int c = lane + 32 * t;
        hr[c] = gelu_exact((v[t] - mu) * rstd * lnw[c] + lnb[c]);
      }
    }
  }
}

// ---------------------------------------------------------------------------
// 4) gate = h1 @ gate_w + gate_b ; segment max via float-bits atomic trick
// ---------------------------------------------------------------------------
__global__ void gate_kernel(const float* __restrict__ h1,
                            const float* __restrict__ gw,
                            const float* __restrict__ gb,
                            const int* __restrict__ batch,
                            float* __restrict__ gate,
                            unsigned* __restrict__ segmax_bits, int N) {
  int row = (blockIdx.x * blockDim.x + threadIdx.x) >> 5;
  int lane = threadIdx.x & 31;
  if (row >= N) return;
  float s = 0.f;
#pragma unroll
  for (int t = 0; t < 4; ++t) {
    int c = lane + 32 * t;
    s += h1[(size_t)row * 128 + c] * gw[c];
  }
  s = wave_sum(s);
  if (lane == 0) {
    s += gb[0];
    gate[row] = s;
    int b = batch[row];
    if (s >= 0.f) atomicMax((int*)(segmax_bits + b), __float_as_int(s));
    else          atomicMin(segmax_bits + b, (unsigned)__float_as_int(s));
  }
}

__global__ void softmax_e_kernel(const float* __restrict__ gate,
                                 const int* __restrict__ batch,
                                 const unsigned* __restrict__ segmax_bits,
                                 float* __restrict__ ebuf,
                                 float* __restrict__ segden, int N) {
  int i = blockIdx.x * blockDim.x + threadIdx.x;
  if (i >= N) return;
  int b = batch[i];
  float m = __int_as_float((int)segmax_bits[b]);
  if (m < -1e30f) m = 0.f;   // empty segment: seg_max -> 0 (isfinite guard)
  float e = expf(gate[i] - m);
  ebuf[i] = e;
  atomicAdd(segden + b, e);
}

__global__ void attn_agg_kernel(const float* __restrict__ h1,
                                const float* __restrict__ ebuf,
                                const float* __restrict__ segden,
                                const int* __restrict__ batch,
                                float* __restrict__ attn_out,
                                float* __restrict__ graphemb, int N) {
  int row = (blockIdx.x * blockDim.x + threadIdx.x) >> 5;
  int lane = threadIdx.x & 31;
  if (row >= N) return;
  int b = batch[row];
  float a = ebuf[row] / segden[b];
  if (lane == 0) attn_out[row] = a;
#pragma unroll
  for (int t = 0; t < 4; ++t) {
    int c = lane + 32 * t;
    atomicAdd(graphemb + (size_t)b * 128 + c, a * h1[(size_t)row * 128 + c]);
  }
}

// ---------------------------------------------------------------------------
// 5) msg_emb = GELU(text[G,768] @ msg_W[768,128] + b)  -- WMMA, K chunked
// ---------------------------------------------------------------------------
#define LDA_MSG 34   // halves; 17-DWORD stride -> conflict-free
__global__ void msg_gemm_kernel(const float* __restrict__ X,
                                const _Float16* __restrict__ Wsw,
                                const float* __restrict__ b,
                                float* __restrict__ Y, int G) {
  __shared__ _Float16 sA[16 * LDA_MSG];
  const int row0 = blockIdx.x * 16;
  const int tid = threadIdx.x, wave = tid >> 5, lane = tid & 31;
  v8f acc = {};
  for (int kt = 0; kt < 768 / 32; ++kt) {
    __syncthreads();
    {
      // 256 threads cover 16 rows x 16 two-half chunks exactly
      int r = tid >> 4, c2 = (tid & 15) * 2;
      int gr = row0 + r;
      float a0 = 0.f, a1 = 0.f;
      if (gr < G) {
        const float* xr = X + (size_t)gr * 768 + kt * 32 + c2;
        a0 = xr[0]; a1 = xr[1];
      }
      union { unsigned u; _Float16 h[2]; } p;
      p.h[0] = (_Float16)a0; p.h[1] = (_Float16)a1;
      *(unsigned*)(sA + r * LDA_MSG + c2) = p.u;
    }
    __syncthreads();
    v16h a  = load_a_frag_lds(sA, LDA_MSG, 0, lane);
    v16h bf = load_b_frag_sw(Wsw, kt, 8, wave, lane);
    acc = __builtin_amdgcn_wmma_f32_16x16x32_f16(false, a, false, bf,
                                                 (short)0, acc, false, false);
  }
  const int col = wave * 16 + (lane & 15);
  const int grp = lane >> 4;
  const float bias = b[col];
#pragma unroll
  for (int v = 0; v < 8; ++v) {
    int r = row0 + v + 8 * grp;
    if (r < G) Y[(size_t)r * 128 + col] = gelu_exact(acc[v] + bias);
  }
}

// ---------------------------------------------------------------------------
// 6) feat_emb = GELU(feats[G,14] @ feat_W[14,128] + b)  -- K=14, scalar
// ---------------------------------------------------------------------------
__global__ void feat_kernel(const float* __restrict__ F,
                            const float* __restrict__ W,
                            const float* __restrict__ b,
                            float* __restrict__ Y, int G) {
  int i = blockIdx.x * blockDim.x + threadIdx.x;
  if (i >= G * 128) return;
  int g = i >> 7, j = i & 127;
  float s = b[j];
#pragma unroll
  for (int k = 0; k < 14; ++k) s += F[g * 14 + k] * W[k * 128 + j];
  Y[i] = gelu_exact(s);
}

// ---------------------------------------------------------------------------
// 7) concat(3x128) -> LN -> @fc1_W + fc1_b ; also emit graph_embedding copy
// ---------------------------------------------------------------------------
__global__ void final_kernel(const float* __restrict__ graphemb,
                             const float* __restrict__ msgemb,
                             const float* __restrict__ featemb,
                             const float* __restrict__ gwscalar,
                             const float* __restrict__ mw,
                             const float* __restrict__ mb,
                             const float* __restrict__ fcW,
                             const float* __restrict__ fcb,
                             float* __restrict__ logits,
                             float* __restrict__ ge_out, int G) {
  int g = (blockIdx.x * blockDim.x + threadIdx.x) >> 5;
  int lane = threadIdx.x & 31;
  if (g >= G) return;
  float gw = gwscalar[0];
  float v[12], s = 0.f;
#pragma unroll
  for (int t = 0; t < 12; ++t) {
    int idx = lane + 32 * t;
    float x;
    if (idx < 128)      x = gw * graphemb[(size_t)g * 128 + idx];
    else if (idx < 256) x = msgemb[(size_t)g * 128 + (idx - 128)];
    else                x = featemb[(size_t)g * 128 + (idx - 256)];
    v[t] = x; s += x;
  }
  s = wave_sum(s);
  float mu = s * (1.f / 384.f);
  float vs = 0.f;
#pragma unroll
  for (int t = 0; t < 12; ++t) { float d = v[t] - mu; vs += d * d; }
  vs = wave_sum(vs);
  float rstd = rsqrtf(vs * (1.f / 384.f) + 1e-5f);
  float acc = 0.f;
#pragma unroll
  for (int t = 0; t < 12; ++t) {
    int idx = lane + 32 * t;
    acc += ((v[t] - mu) * rstd * mw[idx] + mb[idx]) * fcW[idx];
  }
  acc = wave_sum(acc);
  if (lane == 0) logits[g] = acc + fcb[0];
#pragma unroll
  for (int t = 0; t < 4; ++t) {
    int c = lane + 32 * t;
    ge_out[(size_t)g * 128 + c] = graphemb[(size_t)g * 128 + c];
  }
}

// ---------------------------------------------------------------------------
// Host launcher
// ---------------------------------------------------------------------------
extern "C" void kernel_launch(void* const* d_in, const int* in_sizes, int n_in,
                              void* d_out, int out_size, void* d_ws, size_t ws_size,
                              hipStream_t stream) {
  const float* x          = (const float*)d_in[0];
  const int*   edge       = (const int*)d_in[1];
  const int*   batch      = (const int*)d_in[2];
  const float* text       = (const float*)d_in[4];
  const float* feats      = (const float*)d_in[5];
  const float* ln_proj_w  = (const float*)d_in[6];
  const float* ln_proj_b  = (const float*)d_in[7];
  const float* W_l        = (const float*)d_in[8];
  const float* b_l        = (const float*)d_in[9];
  const float* W_r        = (const float*)d_in[10];
  const float* ln_conv_w  = (const float*)d_in[11];
  const float* ln_conv_b  = (const float*)d_in[12];
  const float* gate_w     = (const float*)d_in[13];
  const float* gate_b     = (const float*)d_in[14];
  const float* graph_wt   = (const float*)d_in[15];
  const float* msg_W      = (const float*)d_in[16];
  const float* msg_b      = (const float*)d_in[17];
  const float* feat_W     = (const float*)d_in[18];
  const float* feat_b     = (const float*)d_in[19];
  const float* mixed_w    = (const float*)d_in[20];
  const float* mixed_b    = (const float*)d_in[21];
  const float* fc1_W      = (const float*)d_in[22];
  const float* fc1_b      = (const float*)d_in[23];

  const int N = in_sizes[0] / 128;
  const int E = in_sizes[1] / 2;
  const int G = in_sizes[4] / 768;
  const int* src = edge;
  const int* dst = edge + E;

  // workspace carve (256B aligned chunks)
  char* wp = (char*)d_ws;
  auto carve = [&](size_t bytes) {
    char* p = wp;
    wp += (bytes + 255) & ~(size_t)255;
    return p;
  };
  _Float16*  h0h       = (_Float16*)carve((size_t)N * 128 * 2);
  _Float16*  aggh      = (_Float16*)carve((size_t)N * 128 * 2);
  float*     h1        = (float*)carve((size_t)N * 128 * 4);
  float*     deg       = (float*)carve((size_t)N * 4);
  float*     gate      = (float*)carve((size_t)N * 4);
  float*     ebuf      = (float*)carve((size_t)N * 4);
  unsigned*  segmax    = (unsigned*)carve((size_t)G * 4);
  float*     segden    = (float*)carve((size_t)G * 4);
  float*     graphemb  = (float*)carve((size_t)G * 128 * 4);
  float*     msgemb    = (float*)carve((size_t)G * 128 * 4);
  float*     featemb   = (float*)carve((size_t)G * 128 * 4);
  _Float16*  Wl_sw     = (_Float16*)carve((size_t)128 * 128 * 2);
  _Float16*  Wr_sw     = (_Float16*)carve((size_t)128 * 128 * 2);
  _Float16*  msgW_sw   = (_Float16*)carve((size_t)768 * 128 * 2);
  (void)ws_size; (void)n_in; (void)out_size;

  float* logits   = (float*)d_out;
  float* ge_out   = logits + G;
  float* attn_out = ge_out + (size_t)G * 128;

  auto blk = [](long long n) { return (int)((n + 255) / 256); };

  // init (aggh zeroed via float view: 0x00000000 == two +0.0h halves)
  fill_kernel<<<blk((long long)N * 64), 256, 0, stream>>>((float*)aggh, 0.f, N * 64);
  fill_kernel<<<blk(N), 256, 0, stream>>>(deg, 0.f, N);
  fill_kernel<<<blk(G), 256, 0, stream>>>(segden, 0.f, G);
  fill_kernel<<<blk((long long)G * 128), 256, 0, stream>>>(graphemb, 0.f, G * 128);
  fill_kernel<<<blk(G), 256, 0, stream>>>((float*)segmax, -INFINITY, G);

  // weight swizzle to WMMA B-fragment order (f16)
  swizzle_b_kernel<<<blk(128 * 128), 256, 0, stream>>>(W_l, Wl_sw, 128, 128);
  swizzle_b_kernel<<<blk(128 * 128), 256, 0, stream>>>(W_r, Wr_sw, 128, 128);
  swizzle_b_kernel<<<blk(768 * 128), 256, 0, stream>>>(msg_W, msgW_sw, 768, 128);

  // pipeline
  ln_x_kernel<<<blk((long long)N * 32), 256, 0, stream>>>(x, ln_proj_w, ln_proj_b, h0h, N);
  scatter_kernel<<<blk((long long)E * 32), 256, 0, stream>>>(h0h, src, dst, aggh, deg, E);
  sage_gemm_kernel<<<(N + 15) / 16, 256, 0, stream>>>(h0h, aggh, deg, Wl_sw, Wr_sw,
                                                      b_l, ln_conv_w, ln_conv_b, h1, N);
  gate_kernel<<<blk((long long)N * 32), 256, 0, stream>>>(h1, gate_w, gate_b, batch,
                                                          gate, segmax, N);
  softmax_e_kernel<<<blk(N), 256, 0, stream>>>(gate, batch, segmax, ebuf, segden, N);
  attn_agg_kernel<<<blk((long long)N * 32), 256, 0, stream>>>(h1, ebuf, segden, batch,
                                                              attn_out, graphemb, N);
  msg_gemm_kernel<<<(G + 15) / 16, 256, 0, stream>>>(text, msgW_sw, msg_b, msgemb, G);
  feat_kernel<<<blk((long long)G * 128), 256, 0, stream>>>(feats, feat_W, feat_b, featemb, G);
  final_kernel<<<blk((long long)G * 32), 256, 0, stream>>>(graphemb, msgemb, featemb,
                                                           graph_wt, mixed_w, mixed_b,
                                                           fc1_W, fc1_b, logits, ge_out, G);
}
